// JointNet_2783138808540
// MI455X (gfx1250) — matrix operand order repaired and checked by reading, not compiled, not run on hardware
//
#include <hip/hip_runtime.h>
#include <hip/hip_bf16.h>
#include <cstdint>
#include <cstddef>

typedef __attribute__((ext_vector_type(16))) _Float16 v16h;
typedef __attribute__((ext_vector_type(8)))  _Float16 v8h;
typedef __attribute__((ext_vector_type(8)))  float    v8f;

// Problem shape (fixed by reference)
constexpr int B = 4, T = 256, U = 96, D = 512, H = 512, V = 1024;
constexpr int BT = B * T;        // 1024 encoder rows
constexpr int BU = B * U;        // 384 decoder rows
constexpr int MROWS = 32;        // output rows per workgroup (2 WMMA A-tiles)
constexpr int UT = U / MROWS;    // 3 u-tiles per (b,t)
constexpr int MTILES = BT * UT;  // 3072 workgroups for the main GEMM
constexpr int LDS_K = H + 8;     // padded f16 row stride (16B pad) -> conflict-free ds_load_b128

__device__ __forceinline__ float gelu_tanh(float x) {
    float x3 = x * x * x;
    float t  = tanhf(0.7978845608028654f * (x + 0.044715f * x3));
    return 0.5f * x * (1.0f + t);
}

// he = enc @ W1[0:D]          (bias == nullptr, koff == 0)
// hdb = dec @ W1[D:2D] + b1   (bias == b1,      koff == D)
// 16 rows per block so the W1 column slab is reused 16x (stays in L2).
__global__ __launch_bounds__(256) void proj_kernel(
    const float* __restrict__ src,   // rows x D
    const float* __restrict__ W1,    // (2D) x H
    const float* __restrict__ bias,  // H or nullptr
    float* __restrict__ dst,         // rows x H
    int koff)
{
    const int h  = blockIdx.y * 256 + threadIdx.x;
    const int r0 = blockIdx.x * 16;

    float acc[16];
#pragma unroll
    for (int r = 0; r < 16; ++r) acc[r] = 0.0f;

    const float* w = W1 + (size_t)koff * H + h;
#pragma unroll 4
    for (int k = 0; k < D; ++k) {
        float wv = w[(size_t)k * H];
#pragma unroll
        for (int r = 0; r < 16; ++r)
            acc[r] = fmaf(src[(size_t)(r0 + r) * D + k], wv, acc[r]);
    }

    float bv = bias ? bias[h] : 0.0f;
#pragma unroll
    for (int r = 0; r < 16; ++r)
        dst[(size_t)(r0 + r) * H + h] = acc[r] + bv;
}

// W2 (H x V, f32, row-major) -> W2T (V x H, f16, row-major) via 64x64 LDS tile
// transpose: coalesced reads along V, coalesced f16 writes along H.
__global__ __launch_bounds__(256) void w2t_kernel(
    const float* __restrict__ W2, _Float16* __restrict__ W2T)
{
    __shared__ float tile[64][65];   // +1 pad: conflict-free transposed reads
    const int vb = blockIdx.x * 64;
    const int kb = blockIdx.y * 64;
    const int tx = threadIdx.x & 63;
    const int ty = threadIdx.x >> 6;  // 0..3

#pragma unroll
    for (int i = 0; i < 64; i += 4)
        tile[ty + i][tx] = W2[(size_t)(kb + ty + i) * V + (vb + tx)];
    __syncthreads();
#pragma unroll
    for (int i = 0; i < 64; i += 4)
        W2T[(size_t)(vb + ty + i) * H + (kb + tx)] = (_Float16)tile[tx][ty + i];
}

// Fused: h = gelu(he[bt] + hdb[b,u]) staged as f16 in LDS (32 rows), then
// 32x1024 GEMM vs W2T with v_wmma_f32_16x16x32_f16. One workgroup = 32-row
// M block; 8 waves x 128 cols; every B-fragment load feeds TWO WMMAs.
__global__ __launch_bounds__(256) void joint_kernel(
    const float* __restrict__ he,     // BT x H
    const float* __restrict__ hdb,    // BU x H  (b1 already folded in)
    const _Float16* __restrict__ W2T, // V x H, f16
    float* __restrict__ out)          // (BT*U) x V
{
    __shared__ _Float16 hA[MROWS * LDS_K];   // 33,280 B

    const int tid = threadIdx.x;
    const int mt  = blockIdx.x;     // M block id
    const int bt  = mt / UT;        // = b*T + t
    const int ut  = mt - bt * UT;
    const int b   = bt / T;
    const int u0  = ut * MROWS;
    const size_t rowbase = (size_t)bt * U + u0;  // first output row of this block

    // ---- Stage 1: build f16 h-block (32 x 512) in LDS: gelu(he + hd + b1) ----
    const float* heRow = he  + (size_t)bt * H;
    const float* hdRow = hdb + ((size_t)b * U + u0) * H;
    for (int i = 0; i < (MROWS * H) / 256; ++i) {
        int e = i * 256 + tid;        // 0..16383
        int r = e >> 9;               // / 512
        int k = e & (H - 1);
        float x = heRow[k] + hdRow[(size_t)r * H + k];
        hA[r * LDS_K + k] = (_Float16)gelu_tanh(x);
    }
    __syncthreads();

    // ---- Stage 2: two 16x128 output tiles per wave; 16 K-steps x 16 WMMAs ----
    const int lane  = tid & 31;
    const int wave  = tid >> 5;
    const int ar    = lane & 15;        // A row = lane%16
    const int ahalf = (lane >> 4) & 1;  // lanes 16..31 take the +8 / +16 K half

    // A fragment (16-bit A 16x32): lanes<16: K 0..15; lanes>=16: K 16..31
    const _Float16* aBase0 = &hA[ar * LDS_K + ahalf * 16];          // rows 0..15
    const _Float16* aBase1 = aBase0 + 16 * LDS_K;                   // rows 16..31

    // B fragment (16-bit B 32x16): lane<16 -> col lane, K 0..15; lane>=16 -> col lane-16, K 16..31
    const int v0 = wave * 128;
    const _Float16* bBase = W2T + (size_t)(v0 + (lane & 15)) * H + ahalf * 16;

    v8f acc0[8] = {};
    v8f acc1[8] = {};

    for (int kc = 0; kc < H; kc += 32) {
        v8h a0lo = *(const v8h*)(aBase0 + kc);
        v8h a0hi = *(const v8h*)(aBase0 + kc + 8);
        v8h a1lo = *(const v8h*)(aBase1 + kc);
        v8h a1hi = *(const v8h*)(aBase1 + kc + 8);
        v16h a0, a1;
#pragma unroll
        for (int i = 0; i < 8; ++i) {
            a0[i] = a0lo[i]; a0[8 + i] = a0hi[i];
            a1[i] = a1lo[i]; a1[8 + i] = a1hi[i];
        }

        __builtin_prefetch(bBase + kc + 64, 0, 0);  // global_prefetch_b8 on next W2T chunk

#pragma unroll
        for (int n = 0; n < 8; ++n) {
            v16h bf = *(const v16h*)(bBase + (size_t)(n * 16) * H + kc);
            acc0[n] = __builtin_amdgcn_wmma_f32_16x16x32_f16(
                false, a0, false, bf, (short)0, acc0[n], false, false);
            acc1[n] = __builtin_amdgcn_wmma_f32_16x16x32_f16(
                false, a1, false, bf, (short)0, acc1[n], false, false);
        }
    }

    // ---- Epilogue: C layout -> VGPR j holds rows j (lanes 0-15) / j+8 (lanes 16-31) ----
    const int mrow = (lane >> 4) ? 8 : 0;
    const int ncol = lane & 15;
#pragma unroll
    for (int n = 0; n < 8; ++n) {
        size_t col = (size_t)v0 + n * 16 + ncol;
        float* op0 = out + (rowbase + mrow) * (size_t)V + col;
        float* op1 = out + (rowbase + 16 + mrow) * (size_t)V + col;
#pragma unroll
        for (int j = 0; j < 8; ++j) {
            op0[(size_t)j * V] = acc0[n][j];
            op1[(size_t)j * V] = acc1[n][j];
        }
    }
}

extern "C" void kernel_launch(void* const* d_in, const int* in_sizes, int n_in,
                              void* d_out, int out_size, void* d_ws, size_t ws_size,
                              hipStream_t stream) {
    const float* enc = (const float*)d_in[0];  // B*T*D
    const float* dec = (const float*)d_in[1];  // B*U*D
    const float* W1  = (const float*)d_in[2];  // 2D*H
    const float* b1  = (const float*)d_in[3];  // H
    const float* W2  = (const float*)d_in[4];  // H*V
    float* out = (float*)d_out;                // B*T*U*V

    // Workspace layout (3.9 MB total): he (f32) | hdb (f32) | W2T (f16)
    float*    he  = (float*)d_ws;
    float*    hdb = he + (size_t)BT * H;
    _Float16* W2T = (_Float16*)(hdb + (size_t)BU * H);  // offset 2,883,584 B (32B aligned)

    proj_kernel<<<dim3(BT / 16, H / 256), 256, 0, stream>>>(enc, W1, nullptr, he, 0);
    proj_kernel<<<dim3(BU / 16, H / 256), 256, 0, stream>>>(dec, W1, b1, hdb, D);
    w2t_kernel<<<dim3(V / 64, H / 64), 256, 0, stream>>>(W2, W2T);
    joint_kernel<<<MTILES, 256, 0, stream>>>(he, hdb, W2T, out);
}
